// GATClassifier_58918361366988
// MI455X (gfx1250) — compile-verified, matrix operand order checked by reading, not run on hardware
//
#include <hip/hip_runtime.h>

// ---------------- problem constants (match setup_inputs) ----------------
#define BATCH 4
#define NNODES 4096
#define NEDGES 131072
#define EA (NEDGES + NNODES)      // edges + self loops
#define FIN 768
#define HEADS 4
#define HID 64
#define HD (HEADS * HID)          // 256
#define BN (BATCH * NNODES)       // 16384
#define NCLS 6

typedef __attribute__((ext_vector_type(16))) __bf16 v16bf;
typedef __attribute__((ext_vector_type(8)))  __bf16 v8bf;
typedef __attribute__((ext_vector_type(8)))  float  v8f;
typedef int v4i_ty __attribute__((vector_size(16)));   // matches builtin proto

#if defined(__has_builtin)
#if __has_builtin(__builtin_amdgcn_global_load_async_to_lds_b128) && \
    __has_builtin(__builtin_amdgcn_s_wait_asynccnt)
#define USE_ASYNC_LDS 1
#endif
#endif

// ---------------- helpers ----------------
__device__ __forceinline__ int enc_f(float f) {
  int i = __float_as_int(f);
  return i ^ ((i >> 31) & 0x7fffffff);   // monotone float -> int map
}
__device__ __forceinline__ float dec_f(int k) {
  return __int_as_float(k ^ ((k >> 31) & 0x7fffffff));
}

// ---------------- elementwise fp32 -> bf16 ----------------
__global__ void cvt_bf16_kernel(const float* __restrict__ src,
                                __bf16* __restrict__ dst, int n) {
  int i = blockIdx.x * blockDim.x + threadIdx.x;
  if (i < n) dst[i] = (__bf16)src[i];
}

// ---------------- WMMA GEMM: C[M,Nc] f32 = A[M,K] bf16 @ B[K,Nc] bf16 ----
// Block = 256 threads (8 waves) computes a 128x64 C macro-tile.
// Wave w owns rows [w*16, w*16+16) x 64 cols = 4 accumulators; one A fragment
// feeds 4 back-to-back WMMAs per K-step. B K-slice staged in LDS pre-swizzled
// into fragment order (ds_load_b128 reads); A K-slice staged via async
// global->LDS copies (gfx1250 GLOBAL_LOAD_ASYNC_TO_LDS_B128 + s_wait_asynccnt).
__global__ __launch_bounds__(256)
void gemm_bf16_wmma(const __bf16* __restrict__ A, const __bf16* __restrict__ B,
                    float* __restrict__ C, int M, int Nc, int K) {
  __shared__ __bf16 smemA[128 * 32];       // 8 KB, linear [row][k]
  __shared__ __bf16 smemB[4 * 32 * 16];    // 4 KB, fragment-order [tile][lane][elem]

  const int colB = Nc >> 6;                // 64-wide column blocks
  const int br = blockIdx.x / colB;
  const int bc = blockIdx.x - br * colB;
  const int rm0 = br * 128;
  const int cn  = bc * 64;

  const int tid  = threadIdx.x;
  const int w    = tid >> 5;
  const int lane = tid & 31;
  const int r    = lane & 15;
  const int half = lane >> 4;
  const int hk   = half << 3;              // 0 or 8

  v8f acc[4];
#pragma unroll
  for (int t = 0; t < 4; ++t)
    acc[t] = (v8f){0.f, 0.f, 0.f, 0.f, 0.f, 0.f, 0.f, 0.f};

  // B staging decomposition (fixed per thread)
  const int bk  = tid >> 3;                // slice-local k row: 0..31
  const int bc0 = (tid & 7) * 8;           // 8 contiguous cols
  const int be  = (bk & 7) + ((bk >> 4) << 3);   // fragment elem index
  const int blh = (bk >> 3) & 1;                 // fragment lane half

  for (int k0 = 0; k0 < K; k0 += 32) {
    __syncthreads();

    // ---- stage A slice (128 x 32) ----
#pragma unroll
    for (int j = 0; j < 2; ++j) {
      int chunk = tid + 256 * j;           // 512 x 16B chunks
      int row = chunk >> 2;
      int ko  = (chunk & 3) * 8;
      const __bf16* gsrc = A + (size_t)(rm0 + row) * K + k0 + ko;
#ifdef USE_ASYNC_LDS
      __builtin_amdgcn_global_load_async_to_lds_b128(
          (__attribute__((address_space(1))) v4i_ty*)gsrc,
          (__attribute__((address_space(3))) v4i_ty*)(&smemA[chunk * 8]), 0, 0);
#else
      v8bf av = *(const v8bf*)gsrc;
      *(v8bf*)(&smemA[chunk * 8]) = av;
#endif
    }

    // ---- stage B slice (32 x 64), swizzled to WMMA fragment order ----
    {
      v8bf bv = *(const v8bf*)(B + (size_t)(k0 + bk) * Nc + cn + bc0);
#pragma unroll
      for (int j = 0; j < 8; ++j) {
        int c  = bc0 + j;
        int t  = c >> 4;
        int ln = (c & 15) + (blh << 4);
        smemB[(t * 32 + ln) * 16 + be] = bv[j];
      }
    }

#ifdef USE_ASYNC_LDS
    __builtin_amdgcn_s_wait_asynccnt(0);
#endif
    __syncthreads();

    // ---- A fragment: two contiguous 16B LDS reads ----
    const __bf16* ap = &smemA[(w * 16 + r) * 32 + hk];
    v8bf alo = *(const v8bf*)ap;
    v8bf ahi = *(const v8bf*)(ap + 16);
    v16bf a;
#pragma unroll
    for (int i = 0; i < 8; ++i) { a[i] = alo[i]; a[8 + i] = ahi[i]; }

    // ---- 4 WMMAs, reusing the A fragment ----
#pragma unroll
    for (int t = 0; t < 4; ++t) {
      const __bf16* bp = &smemB[(t * 32 + lane) * 16];
      v8bf blo = *(const v8bf*)bp;
      v8bf bhi = *(const v8bf*)(bp + 8);
      v16bf b;
#pragma unroll
      for (int i = 0; i < 8; ++i) { b[i] = blo[i]; b[8 + i] = bhi[i]; }
      acc[t] = __builtin_amdgcn_wmma_f32_16x16x32_bf16(
          false, a, false, b, (short)0, acc[t], false, false);
    }
  }

  // ---- store: VGPR i -> M = 8*half + i, N = lane&15 (per 16x16 tile) ----
#pragma unroll
  for (int t = 0; t < 4; ++t) {
    float* crow =
        C + (size_t)(rm0 + w * 16 + (half << 3)) * Nc + cn + t * 16 + r;
#pragma unroll
    for (int i = 0; i < 8; ++i) crow[(size_t)i * Nc] = acc[t][i];
  }
}

// ---------------- attention score dots: a_s[n,h], a_d[n,h] ----------------
__global__ void attn_scores_kernel(const float* __restrict__ h,
                                   const float* __restrict__ att_s,
                                   const float* __restrict__ att_d,
                                   float* __restrict__ as_o,
                                   float* __restrict__ ad_o,
                                   int BNv, int H, int D) {
  int i = blockIdx.x * blockDim.x + threadIdx.x;
  if (i >= BNv * H) return;
  int n = i / H, hh = i - n * H;
  const float* hp = h + (size_t)n * H * D + (size_t)hh * D;
  const float* sp = att_s + hh * D;
  const float* dp = att_d + hh * D;
  float s = 0.f, d = 0.f;
  for (int k = 0; k < D; ++k) { float v = hp[k]; s += v * sp[k]; d += v * dp[k]; }
  as_o[i] = s; ad_o[i] = d;
}

// ---------------- init m(enc -inf), denom(0), out(0) ----------------
__global__ void init_buffers_kernel(int* __restrict__ m_enc,
                                    float* __restrict__ denom,
                                    float* __restrict__ out, int nm, int nout) {
  int i = blockIdx.x * blockDim.x + threadIdx.x;
  if (i < nm) { m_enc[i] = (int)0x807FFFFFu; denom[i] = 0.f; }
  if (i < nout) out[i] = 0.f;
}

// ---------------- edge pass 1: leaky_relu score + segment max ----------------
__global__ void edge_max_kernel(const int* __restrict__ ei,
                                const float* __restrict__ as_o,
                                const float* __restrict__ ad_o,
                                float* __restrict__ e_buf,
                                int* __restrict__ m_enc, int H) {
  int i = blockIdx.x * blockDim.x + threadIdx.x;
  if (i >= BATCH * EA * H) return;
  int hh = i % H; int t = i / H; int e = t % EA; int b = t / EA;
  int src, dst;
  if (e < NEDGES) {
    const int* p = ei + (size_t)b * 2 * NEDGES;
    src = p[e]; dst = p[NEDGES + e];
  } else {
    src = dst = e - NEDGES;
  }
  float v = as_o[(size_t)(b * NNODES + src) * H + hh] +
            ad_o[(size_t)(b * NNODES + dst) * H + hh];
  v = v > 0.f ? v : 0.2f * v;   // leaky_relu, PyG default slope
  e_buf[i] = v;
  atomicMax(&m_enc[(size_t)(b * NNODES + dst) * H + hh], enc_f(v));
}

// ---------------- edge pass 2: exp(e - m) + segment sum ----------------
__global__ void edge_expsum_kernel(const int* __restrict__ ei,
                                   const int* __restrict__ m_enc,
                                   float* __restrict__ e_buf,
                                   float* __restrict__ denom, int H) {
  int i = blockIdx.x * blockDim.x + threadIdx.x;
  if (i >= BATCH * EA * H) return;
  int hh = i % H; int t = i / H; int e = t % EA; int b = t / EA;
  int dst;
  if (e < NEDGES) dst = ei[(size_t)b * 2 * NEDGES + NEDGES + e];
  else dst = e - NEDGES;
  float m = dec_f(m_enc[(size_t)(b * NNODES + dst) * H + hh]);
  float ex = __expf(e_buf[i] - m);
  e_buf[i] = ex;
  atomicAdd(&denom[(size_t)(b * NNODES + dst) * H + hh], ex);
}

// ---------------- edge pass 3: out[dst] += alpha * h[src] ----------------
__global__ void edge_aggregate_kernel(const int* __restrict__ ei,
                                      const float* __restrict__ e_buf,
                                      const float* __restrict__ denom,
                                      const float* __restrict__ hfeat,
                                      float* __restrict__ out, int H, int D) {
  int i = blockIdx.x * blockDim.x + threadIdx.x;
  int Dg = D >> 2;                      // float4 groups
  if (i >= BATCH * EA * H * Dg) return;
  int g = i % Dg; int t = i / Dg;
  int hh = t % H; t /= H;
  int e = t % EA; int b = t / EA;
  int src, dst;
  if (e < NEDGES) {
    const int* p = ei + (size_t)b * 2 * NEDGES;
    src = p[e]; dst = p[NEDGES + e];
  } else {
    src = dst = e - NEDGES;
  }
  float alpha = e_buf[((size_t)(b * EA + e)) * H + hh] /
                (denom[(size_t)(b * NNODES + dst) * H + hh] + 1e-16f);
  const float4 hv = *(const float4*)(hfeat +
      ((size_t)(b * NNODES + src) * H + hh) * D + g * 4);
  float* op = out + ((size_t)(b * NNODES + dst) * H + hh) * D + g * 4;
  atomicAdd(op + 0, alpha * hv.x);
  atomicAdd(op + 1, alpha * hv.y);
  atomicAdd(op + 2, alpha * hv.z);
  atomicAdd(op + 3, alpha * hv.w);
}

// ---------------- finalize: bias (+ELU), emit bf16 / f32 ----------------
__global__ void finalize_kernel(const float* __restrict__ out,
                                const float* __restrict__ bias,
                                __bf16* __restrict__ next_bf,
                                float* __restrict__ next_f32,
                                int BNv, int HDv, int do_elu) {
  int i = blockIdx.x * blockDim.x + threadIdx.x;
  if (i >= BNv * HDv) return;
  float v = out[i] + bias[i % HDv];
  if (do_elu) v = v > 0.f ? v : (__expf(v) - 1.f);
  if (next_bf)  next_bf[i]  = (__bf16)v;
  if (next_f32) next_f32[i] = v;
}

// ---------------- graph mean readout: [B,N,64] -> [B,64] ----------------
__global__ void graph_mean_kernel(const float* __restrict__ h3,
                                  float* __restrict__ embed) {
  __shared__ float s[256];
  int b = blockIdx.x, tid = threadIdx.x;
  int d = tid & 63, seg = tid >> 6;
  float acc = 0.f;
  for (int n = seg; n < NNODES; n += 4)
    acc += h3[((size_t)b * NNODES + n) * HID + d];
  s[tid] = acc;
  __syncthreads();
  if (seg == 0)
    embed[b * HID + d] =
        (s[d] + s[64 + d] + s[128 + d] + s[192 + d]) / (float)NNODES;
}

// ---------------- classifier + log_softmax loss + argmax ----------------
__global__ void classify_kernel(const float* __restrict__ embed,
                                const float* __restrict__ Wc,
                                const float* __restrict__ bc,
                                const int* __restrict__ labels,
                                float* __restrict__ outp) {
  __shared__ float lg[32];
  int t = threadIdx.x;
  if (t < BATCH * NCLS) {
    int b = t / NCLS, c = t - b * NCLS;
    float v = bc[c];
    for (int k = 0; k < HID; ++k) v += embed[b * HID + k] * Wc[k * NCLS + c];
    lg[t] = v;
  }
  __syncthreads();
  if (t == 0) {
    float loss = 0.f;
    for (int b = 0; b < BATCH; ++b) {
      float mx = lg[b * NCLS]; int am = 0;
      for (int c = 1; c < NCLS; ++c)
        if (lg[b * NCLS + c] > mx) { mx = lg[b * NCLS + c]; am = c; }
      float se = 0.f;
      for (int c = 0; c < NCLS; ++c) se += __expf(lg[b * NCLS + c] - mx);
      float lse = mx + __logf(se);
      int lab = labels[b];
      loss += (lse - lg[b * NCLS + lab]);
      outp[b] = (float)am;           // pred
      outp[BATCH + b] = (float)lab;  // labels passthrough
    }
    outp[2 * BATCH] = loss / (float)BATCH;  // mean NLL
  }
}

// ---------------- host side ----------------
static inline int gridFor(long n, int blk) { return (int)((n + blk - 1) / blk); }

static void run_gat_layer(const __bf16* Xbf, const __bf16* Wbf,
                          const float* att_s, const float* att_d,
                          const float* bias, int K, int H, int D, int do_elu,
                          float* hbuf, float* as_buf, float* ad_buf,
                          int* m_enc, float* denom, float* e_buf, float* outbuf,
                          __bf16* next_bf, float* next_f32, const int* ei,
                          hipStream_t stream) {
  const int Nc = H * D;
  // 1) h = X @ W  (WMMA, 128x64 macro-tiles)
  int blocks = (BN / 128) * (Nc / 64);
  gemm_bf16_wmma<<<blocks, 256, 0, stream>>>(Xbf, Wbf, hbuf, BN, Nc, K);
  // 2) attention dots
  attn_scores_kernel<<<gridFor((long)BN * H, 256), 256, 0, stream>>>(
      hbuf, att_s, att_d, as_buf, ad_buf, BN, H, D);
  // 3) init reducers + output accumulator
  long nm = (long)BATCH * NNODES * H, nout = (long)BN * Nc;
  init_buffers_kernel<<<gridFor(nout > nm ? nout : nm, 256), 256, 0, stream>>>(
      m_enc, denom, outbuf, (int)nm, (int)nout);
  // 4) edge softmax (max, exp-sum) + aggregate
  long ne = (long)BATCH * EA * H;
  edge_max_kernel<<<gridFor(ne, 256), 256, 0, stream>>>(ei, as_buf, ad_buf,
                                                        e_buf, m_enc, H);
  edge_expsum_kernel<<<gridFor(ne, 256), 256, 0, stream>>>(ei, m_enc, e_buf,
                                                           denom, H);
  edge_aggregate_kernel<<<gridFor(ne * (D / 4), 256), 256, 0, stream>>>(
      ei, e_buf, denom, hbuf, outbuf, H, D);
  // 5) bias + (ELU) + emit next-layer inputs
  finalize_kernel<<<gridFor(nout, 256), 256, 0, stream>>>(
      outbuf, bias, next_bf, next_f32, BN, Nc, do_elu);
}

extern "C" void kernel_launch(void* const* d_in, const int* in_sizes, int n_in,
                              void* d_out, int out_size, void* d_ws,
                              size_t ws_size, hipStream_t stream) {
  const float* x    = (const float*)d_in[0];
  const int* labels = (const int*)d_in[1];
  const int* ei     = (const int*)d_in[2];
  const float* W1  = (const float*)d_in[3];
  const float* as1 = (const float*)d_in[4];
  const float* ad1 = (const float*)d_in[5];
  const float* b1  = (const float*)d_in[6];
  const float* W2  = (const float*)d_in[7];
  const float* as2 = (const float*)d_in[8];
  const float* ad2 = (const float*)d_in[9];
  const float* b2  = (const float*)d_in[10];
  const float* W3  = (const float*)d_in[11];
  const float* as3 = (const float*)d_in[12];
  const float* ad3 = (const float*)d_in[13];
  const float* b3  = (const float*)d_in[14];
  const float* Wc  = (const float*)d_in[15];
  const float* bc  = (const float*)d_in[16];
  float* out = (float*)d_out;

  // carve workspace
  size_t off = 0;
  auto carve = [&](size_t bytes) -> void* {
    off = (off + 255) & ~(size_t)255;
    void* p = (char*)d_ws + off;
    off += bytes;
    return p;
  };
  __bf16* Xbf   = (__bf16*)carve((size_t)BN * FIN * 2);
  __bf16* W1bf  = (__bf16*)carve((size_t)FIN * HD * 2);
  __bf16* W2bf  = (__bf16*)carve((size_t)HD * HD * 2);
  __bf16* W3bf  = (__bf16*)carve((size_t)HD * HID * 2);
  float*  hbuf  = (float*)carve((size_t)BN * HD * 4);
  float*  outb  = (float*)carve((size_t)BN * HD * 4);
  __bf16* nxbf  = (__bf16*)carve((size_t)BN * HD * 2);
  float*  h3buf = (float*)carve((size_t)BN * HID * 4);
  float*  as_b  = (float*)carve((size_t)BN * HEADS * 4);
  float*  ad_b  = (float*)carve((size_t)BN * HEADS * 4);
  int*    m_enc = (int*)carve((size_t)BN * HEADS * 4);
  float*  denom = (float*)carve((size_t)BN * HEADS * 4);
  float*  e_buf = (float*)carve((size_t)BATCH * EA * HEADS * 4);
  float*  embed = (float*)carve((size_t)BATCH * HID * 4);

  // fp32 -> bf16 conversions
  cvt_bf16_kernel<<<gridFor((long)BN * FIN, 256), 256, 0, stream>>>(x, Xbf, BN * FIN);
  cvt_bf16_kernel<<<gridFor((long)FIN * HD, 256), 256, 0, stream>>>(W1, W1bf, FIN * HD);
  cvt_bf16_kernel<<<gridFor((long)HD * HD, 256), 256, 0, stream>>>(W2, W2bf, HD * HD);
  cvt_bf16_kernel<<<gridFor((long)HD * HID, 256), 256, 0, stream>>>(W3, W3bf, HD * HID);

  // layer 1: 768 -> 4x64, concat, ELU
  run_gat_layer(Xbf, W1bf, as1, ad1, b1, FIN, HEADS, HID, 1,
                hbuf, as_b, ad_b, m_enc, denom, e_buf, outb,
                nxbf, nullptr, ei, stream);
  // layer 2: 256 -> 4x64, concat, ELU
  run_gat_layer(nxbf, W2bf, as2, ad2, b2, HD, HEADS, HID, 1,
                hbuf, as_b, ad_b, m_enc, denom, e_buf, outb,
                nxbf, nullptr, ei, stream);
  // layer 3: 256 -> 1x64, mean over 1 head == identity, no ELU
  run_gat_layer(nxbf, W3bf, as3, ad3, b3, HD, 1, HID, 0,
                hbuf, as_b, ad_b, m_enc, denom, e_buf, outb,
                nullptr, h3buf, ei, stream);

  // readout + classifier + loss
  graph_mean_kernel<<<BATCH, 256, 0, stream>>>(h3buf, embed);
  classify_kernel<<<1, 32, 0, stream>>>(embed, Wc, bc, labels, out);
}
